// Scalars_DitGaussian_dynamics_6949257084992
// MI455X (gfx1250) — compile-verified, matrix operand order checked by reading, not run on hardware
//
#include <hip/hip_runtime.h>
#include <math.h>

// ---------------------------------------------------------------------------
// Types for CDNA5 WMMA (wave32): v_wmma_f32_16x16x32_f16
// ---------------------------------------------------------------------------
typedef _Float16 half16 __attribute__((ext_vector_type(16)));
typedef _Float16 h8v    __attribute__((ext_vector_type(8)));
typedef float    float8 __attribute__((ext_vector_type(8)));
typedef float    flt4   __attribute__((ext_vector_type(4)));
typedef unsigned int u32x4 __attribute__((ext_vector_type(4)));
typedef int          i32x8 __attribute__((ext_vector_type(8)));
typedef int          i32x4 __attribute__((ext_vector_type(4)));

#if defined(__gfx1250__) && __has_builtin(__builtin_amdgcn_tensor_load_to_lds) && \
    __has_builtin(__builtin_amdgcn_s_wait_tensorcnt)
#define USE_TDM 1
#else
#define USE_TDM 0
#endif

#define BSZ   16
#define NND   128
#define ROWS  (BSZ*NND)   // 2048
#define HIDD  256

__device__ __forceinline__ float geluf(float x) {
    return 0.5f * x * (1.0f + erff(x * 0.70710678118654752f));
}
__device__ __forceinline__ float siluf(float x) {
    return x / (1.0f + expf(-x));
}

// ---------------------------------------------------------------------------
// WMMA GEMM: C[M,N] = epilogue(A[M,K] @ B[K,N]).
// Requirements: M%16==0, K%32==0, N%16==0 (ragged shapes are zero-padded).
// Block = 128 threads (4 waves). Block tile = 16 x 64 (one 16x16 per wave).
// A tile (16x32 f32) staged by the Tensor Data Mover (TDM) when available;
// B tile (32x64) staged transposed as f16 by VALU. Fragments built with 16B
// ds reads per the CDNA5 16-bit operand layout:
//   lane l: m/n = l&15, hi = l>=16; element e: k = (e&7) + hi*8 + (e>>3)*16
// ---------------------------------------------------------------------------
__global__ __launch_bounds__(128) void wmma_gemm_kernel(
    const float* __restrict__ A, int lda,
    const float* __restrict__ B, int ldb,
    const float* __restrict__ bias, float bmult,
    float* __restrict__ C, int ldc,
    int M, int N, int K, int act,
    const float* __restrict__ rowscale,
    const float* __restrict__ rowmask,
    const float* __restrict__ gate, int gld,
    const float* __restrict__ residual, int ldr)
{
    __shared__ float    aS32[16 * 32];  // aS32[m*32 + k]  (TDM destination)
    __shared__ _Float16 aS[16 * 32];    // fallback f16 A tile
    __shared__ _Float16 bT[64 * 32];    // bT[c*32 + k]   (transposed)

    const int tm  = blockIdx.y * 16;
    const int tn0 = blockIdx.x * 64;
    const int tid  = threadIdx.x;
    const int wave = tid >> 5;
    const int lane = tid & 31;
    const int hi   = lane >> 4;
    const int lm   = lane & 15;
    const int myn  = tn0 + wave * 16;
    const bool nactive = (myn < N);

    // staging roles
    const int am = tid >> 3;          // 0..15 : A row
    const int ag = (tid & 7) * 4;     // 0..28 : A k-chunk (4 floats)
    const int br = tid >> 2;          // 0..31 : B k-row
    const int bc0 = tn0 + (tid & 3) * 16;   // 16-col group
    const bool bvalid = (bc0 < N);

    float8 acc = {0.f,0.f,0.f,0.f,0.f,0.f,0.f,0.f};

    const float* __restrict__ Atile = A + (size_t)tm * lda;

#if USE_TDM
    // D# group1: static across the K loop (2D tile: 32 x 16, f32, stride lda)
    i32x8 g1;
    g1[0] = 0x00020000;                         // data_size = 4B (code 2)
    g1[1] = (32 & 0xFFFF) << 16;                // tensor_dim0 = 32  [63:48]
    g1[2] = (16 & 0xFFFF) << 16;                // tensor_dim1 = 16  [111:80]
    g1[3] = (32 << 16);                         // tile_dim0 = 32    [127:112]
    g1[4] = 16;                                 // tile_dim1 = 16, tile_dim2 = 0
    g1[5] = lda;                                // tensor_dim0_stride[31:0]
    g1[6] = 0;                                  // stride0[47:32] | stride1[15:0]
    g1[7] = 0;                                  // stride1[47:16]
    const i32x4 gz  = {0, 0, 0, 0};
    const i32x8 gz8 = {0, 0, 0, 0, 0, 0, 0, 0};
    const unsigned aLds = (unsigned)(size_t)(&aS32[0]);   // LDS byte offset
#endif

    for (int kk = 0; kk < K; kk += 32) {
#if USE_TDM
        // ---- stage A (16x32 f32) via Tensor Data Mover ----
        if (wave == 0) {
            unsigned long long ga = (unsigned long long)(size_t)(Atile + kk);
            u32x4 g0;
            g0[0] = 1u;                                        // count=1, user
            g0[1] = aLds;                                      // lds_addr
            g0[2] = (unsigned)ga;                              // global_addr lo
            g0[3] = ((unsigned)(ga >> 32) & 0x01FFFFFFu) | (2u << 30); // hi|type=2
            __builtin_amdgcn_tensor_load_to_lds(g0, g1, gz, gz, gz8, 0);
            __builtin_amdgcn_s_wait_tensorcnt(0);
        }
#else
        // ---- stage A (16x32) f32->f16 via VALU ----
        {
            const float* __restrict__ Arow = Atile + (size_t)am * lda;
            __builtin_prefetch(Arow + kk + 32, 0, 3);
            flt4 av = *(const flt4*)(Arow + kk + ag);
            _Float16* ad = aS + am * 32 + ag;
            ad[0] = (_Float16)av[0];
            ad[1] = (_Float16)av[1];
            ad[2] = (_Float16)av[2];
            ad[3] = (_Float16)av[3];
        }
#endif
        // ---- stage B (32x64 -> transposed f16) ----
        if (bvalid) {
            const float* __restrict__ Brow = B + (size_t)(kk + br) * ldb + bc0;
            __builtin_prefetch(Brow + (size_t)32 * ldb, 0, 3);  // next K tile
            _Float16* bd = bT + ((tid & 3) * 16) * 32 + br;
#pragma unroll
            for (int cc = 0; cc < 16; cc += 4) {
                flt4 bv = *(const flt4*)(Brow + cc);
                bd[(cc + 0) * 32] = (_Float16)bv[0];
                bd[(cc + 1) * 32] = (_Float16)bv[1];
                bd[(cc + 2) * 32] = (_Float16)bv[2];
                bd[(cc + 3) * 32] = (_Float16)bv[3];
            }
        } else {
            _Float16* bd = bT + ((tid & 3) * 16) * 32 + br;
#pragma unroll
            for (int cc = 0; cc < 16; ++cc) bd[cc * 32] = (_Float16)0.f;
        }
        __syncthreads();

        if (nactive) {
            half16 af;
#if USE_TDM
            flt4 x0 = *(const flt4*)(aS32 + lm * 32 + hi * 8);
            flt4 x1 = *(const flt4*)(aS32 + lm * 32 + hi * 8 + 4);
            flt4 x2 = *(const flt4*)(aS32 + lm * 32 + 16 + hi * 8);
            flt4 x3 = *(const flt4*)(aS32 + lm * 32 + 16 + hi * 8 + 4);
#pragma unroll
            for (int e = 0; e < 4; ++e) {
                af[e]      = (_Float16)x0[e];
                af[e + 4]  = (_Float16)x1[e];
                af[e + 8]  = (_Float16)x2[e];
                af[e + 12] = (_Float16)x3[e];
            }
#else
            h8v a0 = *(const h8v*)(aS + lm * 32 + hi * 8);
            h8v a1 = *(const h8v*)(aS + lm * 32 + 16 + hi * 8);
            af = __builtin_shufflevector(a0, a1, 0,1,2,3,4,5,6,7,8,9,10,11,12,13,14,15);
#endif
            h8v b0 = *(const h8v*)(bT + (wave * 16 + lm) * 32 + hi * 8);
            h8v b1 = *(const h8v*)(bT + (wave * 16 + lm) * 32 + 16 + hi * 8);
            half16 bf = __builtin_shufflevector(b0, b1, 0,1,2,3,4,5,6,7,8,9,10,11,12,13,14,15);
            acc = __builtin_amdgcn_wmma_f32_16x16x32_f16(
                false, af, false, bf, (short)0, acc, false, false);
        }
        __syncthreads();
    }

    if (nactive) {
#pragma unroll
        for (int r = 0; r < 8; ++r) {
            int row = tm + r + hi * 8;
            int col = myn + lm;
            float v = acc[r];
            if (bias)     v += bias[col] * bmult;
            if (act == 1) v = geluf(v);
            else if (act == 2) v = siluf(v);
            if (rowscale) v *= rowscale[row];
            if (gate)     v *= gate[(size_t)(row >> 7) * gld + col];
            if (residual) v += residual[(size_t)row * ldr + col];
            if (rowmask)  v *= rowmask[row];
            C[(size_t)row * ldc + col] = v;
        }
    }
}

// ---------------------------------------------------------------------------
// Fused attention per (batch, head): S=QK^T (WMMA) -> masked softmax in LDS
// (f32 -> f16 in place) -> O=P@V (WMMA; P and staged V read from LDS).
// qkv layout: [row=b*128+n][768] = [q(256)|k(256)|v(256)], head h at h*32.
// grid = BS*HEADS, block = 128 (4 waves), dyn LDS = 64KB scores + 8KB V.
// ---------------------------------------------------------------------------
__global__ __launch_bounds__(128) void attn_kernel(
    const float* __restrict__ qkv,
    const float* __restrict__ nmask,
    float* __restrict__ o)
{
    extern __shared__ char smem[];
    float*    sc = (float*)smem;                          // 128x128 f32
    _Float16* vT = (_Float16*)(smem + 128 * 128 * 4);     // vT[c*128 + n]

    const int b = blockIdx.x >> 3;
    const int h = blockIdx.x & 7;
    const int tid  = threadIdx.x;
    const int wave = tid >> 5;
    const int lane = tid & 31;
    const int hi   = lane >> 4;
    const int lm   = lane & 15;

    const float* __restrict__ base = qkv + (size_t)b * NND * 768;
    const float* __restrict__ Q  = base + h * 32;
    const float* __restrict__ Kp = base + 256 + h * 32;
    const float* __restrict__ V  = base + 512 + h * 32;
    const float scale = 0.17677669529663687f;   // 1/sqrt(32)

    // ---- stage V transposed (f16): thread tid handles key row tid ----
    {
        const float* __restrict__ Vrow = V + (size_t)tid * 768;
#pragma unroll
        for (int c4 = 0; c4 < 32; c4 += 4) {
            flt4 v4 = *(const flt4*)(Vrow + c4);
            vT[(c4 + 0) * 128 + tid] = (_Float16)v4[0];
            vT[(c4 + 1) * 128 + tid] = (_Float16)v4[1];
            vT[(c4 + 2) * 128 + tid] = (_Float16)v4[2];
            vT[(c4 + 3) * 128 + tid] = (_Float16)v4[3];
        }
    }

    // ---- scores: 8x8 tiles of 16x16, K=32 (Q and K rows are contiguous) ----
    for (int t = 0; t < 2; ++t) {
        int tmi = wave * 2 + t;
        const float* __restrict__ Qr = Q + (size_t)(tmi * 16 + lm) * 768;
        flt4 q0 = *(const flt4*)(Qr + hi * 8);
        flt4 q1 = *(const flt4*)(Qr + hi * 8 + 4);
        flt4 q2 = *(const flt4*)(Qr + 16 + hi * 8);
        flt4 q3 = *(const flt4*)(Qr + 16 + hi * 8 + 4);
        half16 af;
#pragma unroll
        for (int e = 0; e < 4; ++e) {
            af[e]      = (_Float16)q0[e];
            af[e + 4]  = (_Float16)q1[e];
            af[e + 8]  = (_Float16)q2[e];
            af[e + 12] = (_Float16)q3[e];
        }
        for (int tni = 0; tni < 8; ++tni) {
            const float* __restrict__ Kr = Kp + (size_t)(tni * 16 + lm) * 768;
            flt4 k0 = *(const flt4*)(Kr + hi * 8);
            flt4 k1 = *(const flt4*)(Kr + hi * 8 + 4);
            flt4 k2 = *(const flt4*)(Kr + 16 + hi * 8);
            flt4 k3 = *(const flt4*)(Kr + 16 + hi * 8 + 4);
            half16 bf;
#pragma unroll
            for (int e = 0; e < 4; ++e) {
                bf[e]      = (_Float16)k0[e];
                bf[e + 4]  = (_Float16)k1[e];
                bf[e + 8]  = (_Float16)k2[e];
                bf[e + 12] = (_Float16)k3[e];
            }
            float8 acc = {0.f,0.f,0.f,0.f,0.f,0.f,0.f,0.f};
            acc = __builtin_amdgcn_wmma_f32_16x16x32_f16(
                false, af, false, bf, (short)0, acc, false, false);
#pragma unroll
            for (int r = 0; r < 8; ++r)
                sc[(tmi * 16 + r + hi * 8) * 128 + tni * 16 + lm] = acc[r] * scale;
        }
    }
    __syncthreads();

    // ---- masked softmax, one row per thread; write f16 P in place ----
    {
        const int i = tid;
        float mx = -3.4e38f;
        for (int j = 0; j < 128; ++j) {
            float bia = (1.0f - nmask[b * NND + j]) * -1.0e9f;
            float v = sc[i * 128 + j] + bia;
            sc[i * 128 + j] = v;
            mx = fmaxf(mx, v);
        }
        float s = 0.f;
        for (int j = 0; j < 128; ++j) s += expf(sc[i * 128 + j] - mx);
        float inv = 1.0f / s;
        _Float16* P = (_Float16*)sc;            // row i at half index i*256
        for (int j = 0; j < 128; ++j) {
            float p = expf(sc[i * 128 + j] - mx) * inv;  // read before overwrite
            P[i * 256 + j] = (_Float16)p;
        }
    }
    __syncthreads();

    // ---- O = P(128x128 f16, LDS) @ V(128x32 f16, LDS transposed) ----
    const _Float16* __restrict__ P = (const _Float16*)sc;
    for (int t = 0; t < 2; ++t) {
        int tmi = wave * 2 + t;
        for (int tni = 0; tni < 2; ++tni) {
            float8 acc = {0.f,0.f,0.f,0.f,0.f,0.f,0.f,0.f};
            for (int kk = 0; kk < 128; kk += 32) {
                h8v p0 = *(const h8v*)(P + (tmi * 16 + lm) * 256 + kk + hi * 8);
                h8v p1 = *(const h8v*)(P + (tmi * 16 + lm) * 256 + kk + 16 + hi * 8);
                h8v v0 = *(const h8v*)(vT + (tni * 16 + lm) * 128 + kk + hi * 8);
                h8v v1 = *(const h8v*)(vT + (tni * 16 + lm) * 128 + kk + 16 + hi * 8);
                half16 af = __builtin_shufflevector(p0, p1, 0,1,2,3,4,5,6,7,8,9,10,11,12,13,14,15);
                half16 bf = __builtin_shufflevector(v0, v1, 0,1,2,3,4,5,6,7,8,9,10,11,12,13,14,15);
                acc = __builtin_amdgcn_wmma_f32_16x16x32_f16(
                    false, af, false, bf, (short)0, acc, false, false);
            }
#pragma unroll
            for (int r = 0; r < 8; ++r)
                o[((size_t)b * NND + tmi * 16 + r + hi * 8) * HIDD
                  + h * 32 + tni * 16 + lm] = acc[r];
        }
    }
}

// ---------------------------------------------------------------------------
// zero-pad copy: dst[drows,dcols] <- src[srows,scols]
// ---------------------------------------------------------------------------
__global__ void pad_kernel(const float* __restrict__ src, int srows, int scols,
                           float* __restrict__ dst, int drows, int dcols)
{
    int i = blockIdx.x * blockDim.x + threadIdx.x;
    int tot = drows * dcols;
    if (i >= tot) return;
    int r = i / dcols, c = i % dcols;
    dst[i] = (r < srows && c < scols) ? src[(size_t)r * scols + c] : 0.f;
}

// ---------------------------------------------------------------------------
// prep: Nn, rowmask, rowinv, centered x
// ---------------------------------------------------------------------------
__global__ void prep_kernel(const float* __restrict__ xh,
                            const float* __restrict__ mask,
                            float* __restrict__ xc,
                            float* __restrict__ rowmask,
                            float* __restrict__ rowinv,
                            float* __restrict__ Nn)
{
    int b = blockIdx.x, n = threadIdx.x, row = b * NND + n;
    __shared__ float red[NND];
    __shared__ float bc[4];
    float m  = mask[row];
    float v0 = xh[(size_t)row * 9 + 0];
    float v1 = xh[(size_t)row * 9 + 1];
    float v2 = xh[(size_t)row * 9 + 2];
    float vals[4] = {m, v0, v1, v2};
    for (int q = 0; q < 4; ++q) {
        red[n] = vals[q]; __syncthreads();
        for (int s = 64; s > 0; s >>= 1) {
            if (n < s) red[n] += red[n + s];
            __syncthreads();
        }
        if (n == 0) bc[q] = red[0];
        __syncthreads();
    }
    float nn = bc[0];
    float inv = 1.0f / nn;
    xc[row * 3 + 0] = v0 - bc[1] * inv * m;
    xc[row * 3 + 1] = v1 - bc[2] * inv * m;
    xc[row * 3 + 2] = v2 - bc[3] * inv * m;
    rowmask[row] = m;
    rowinv[row]  = inv;
    if (n == 0) Nn[b] = nn;
}

// ---------------------------------------------------------------------------
// Fused gaussian layer reduction: S[row,k] = sum_j gauss(dist(i,j),k)*pairmask
// ---------------------------------------------------------------------------
__global__ void gauss_kernel(const float* __restrict__ xc,
                             const float* __restrict__ rowmask,
                             const float* __restrict__ g1m, const float* __restrict__ g1s,
                             const float* __restrict__ gkm, const float* __restrict__ gks,
                             float* __restrict__ S1, float* __restrict__ Sk)
{
    int row = blockIdx.x, b = row >> 7;
    __shared__ float dist[NND];
    __shared__ float pm[NND];
    int j = threadIdx.x;
    float dx = xc[(size_t)row * 3 + 0] - xc[((size_t)b * NND + j) * 3 + 0];
    float dy = xc[(size_t)row * 3 + 1] - xc[((size_t)b * NND + j) * 3 + 1];
    float dz = xc[(size_t)row * 3 + 2] - xc[((size_t)b * NND + j) * 3 + 2];
    dist[j] = sqrtf(dx * dx + dy * dy + dz * dz + 1e-12f);
    pm[j] = rowmask[row] * rowmask[b * NND + j];
    __syncthreads();
    int k = threadIdx.x;
    float m1 = g1m[k], s1 = fabsf(g1s[k]) + 1e-5f;
    float mk = gkm[k], sk = fabsf(gks[k]) + 1e-5f;
    float c1 = 0.3989422804014327f / s1;
    float ck = 0.3989422804014327f / sk;
    float a1 = 0.f, akk = 0.f;
    for (int jj = 0; jj < NND; ++jj) {
        float d = dist[jj], p = pm[jj];
        float z1 = (d - m1) / s1;
        float zk = (d - mk) / sk;
        a1  += expf(-0.5f * z1 * z1) * p;
        akk += expf(-0.5f * zk * zk) * p;
    }
    S1[(size_t)row * 128 + k] = a1 * c1;
    Sk[(size_t)row * 128 + k] = akk * ck;
}

// ---------------------------------------------------------------------------
// timestep embedding
// ---------------------------------------------------------------------------
__global__ void temb_kernel(const float* __restrict__ t, float* __restrict__ e)
{
    int b = blockIdx.x, k = threadIdx.x;
    int half = k & 127;
    float fr = expf(-9.210340371976184f * (float)half / 128.0f);
    float a = t[b] * fr;
    e[b * 256 + k] = (k < 128) ? cosf(a) : sinf(a);
}

__global__ void elem_silu_kernel(const float* __restrict__ in, float* __restrict__ out, int n)
{
    int i = blockIdx.x * blockDim.x + threadIdx.x;
    if (i < n) out[i] = siluf(in[i]);
}

// ---------------------------------------------------------------------------
// concat builders (padded leading dims, zero fill)
// ---------------------------------------------------------------------------
__global__ void fcat_kernel(const float* __restrict__ f1, const float* __restrict__ noise1,
                            const float* __restrict__ tvec, const float* __restrict__ rowmask,
                            float* __restrict__ fcat)   // ld 352
{
    int row = blockIdx.x, b = row >> 7;
    float m = rowmask[row];
    for (int c = threadIdx.x; c < 352; c += blockDim.x) {
        float v;
        if (c < 64)       v = f1[(size_t)row * 64 + c];
        else if (c < 80)  v = m * noise1[(size_t)row * 16 + (c - 64)];
        else if (c < 336) v = m * tvec[b * 256 + (c - 80)];
        else              v = 0.f;
        fcat[(size_t)row * 352 + c] = v;
    }
}

__global__ void gxcat_kernel(const float* __restrict__ xc, const float* __restrict__ gamma1,
                             const float* __restrict__ noise0, const float* __restrict__ tvec,
                             const float* __restrict__ rowmask, float* __restrict__ gx) // ld 288
{
    int row = blockIdx.x, b = row >> 7;
    float m = rowmask[row];
    float x0 = xc[row * 3 + 0], x1 = xc[row * 3 + 1], x2 = xc[row * 3 + 2];
    const float* g = gamma1 + b * 9;
    for (int c = threadIdx.x; c < 288; c += blockDim.x) {
        float v;
        if (c < 3)        v = x0 * g[0 * 3 + c] + x1 * g[1 * 3 + c] + x2 * g[2 * 3 + c];
        else if (c < 19)  v = m * noise0[(size_t)row * 16 + (c - 3)];
        else if (c < 275) v = m * tvec[b * 256 + (c - 19)];
        else              v = 0.f;
        gx[(size_t)row * 288 + c] = v;
    }
}

__global__ void cxh_kernel(const float* __restrict__ xc, const float* __restrict__ gamma,
                           const float* __restrict__ xh, float* __restrict__ cxh) // ld 32
{
    int row = blockIdx.x, b = row >> 7;
    float x0 = xc[row * 3 + 0], x1 = xc[row * 3 + 1], x2 = xc[row * 3 + 2];
    const float* g = gamma + b * 9;
    for (int c = threadIdx.x; c < 32; c += blockDim.x) {
        float v;
        if (c < 3)      v = x0 * g[0 * 3 + c] + x1 * g[1 * 3 + c] + x2 * g[2 * 3 + c];
        else if (c < 9) v = xh[(size_t)row * 9 + c];
        else            v = 0.f;
        cxh[(size_t)row * 32 + c] = v;
    }
}

// ---------------------------------------------------------------------------
// column mean over nodes: out[b,c] = sum_n enc[b,n,c] / Nn[b]
// ---------------------------------------------------------------------------
__global__ void colmean_kernel(const float* __restrict__ enc, const float* __restrict__ Nn,
                               float* __restrict__ out)
{
    int b = blockIdx.x, c = threadIdx.x;
    float s = 0.f;
    for (int n = 0; n < NND; ++n) s += enc[((size_t)b * NND + n) * 256 + c];
    out[b * 256 + c] = s / Nn[b];
}

// ---------------------------------------------------------------------------
// 3x3 modified Gram-Schmidt QR (Q only)
// ---------------------------------------------------------------------------
__device__ void qr3(const float* A, float* Q)
{
    float c0[3] = {A[0], A[3], A[6]};
    float c1[3] = {A[1], A[4], A[7]};
    float c2[3] = {A[2], A[5], A[8]};
    float n0 = rsqrtf(c0[0]*c0[0] + c0[1]*c0[1] + c0[2]*c0[2] + 1e-12f);
    float q0[3] = {c0[0]*n0, c0[1]*n0, c0[2]*n0};
    float r01 = q0[0]*c1[0] + q0[1]*c1[1] + q0[2]*c1[2];
    float u1[3] = {c1[0]-r01*q0[0], c1[1]-r01*q0[1], c1[2]-r01*q0[2]};
    float n1 = rsqrtf(u1[0]*u1[0] + u1[1]*u1[1] + u1[2]*u1[2] + 1e-12f);
    float q1[3] = {u1[0]*n1, u1[1]*n1, u1[2]*n1};
    float r02 = q0[0]*c2[0] + q0[1]*c2[1] + q0[2]*c2[2];
    float r12 = q1[0]*c2[0] + q1[1]*c2[1] + q1[2]*c2[2];
    float u2[3] = {c2[0]-r02*q0[0]-r12*q1[0], c2[1]-r02*q0[1]-r12*q1[1], c2[2]-r02*q0[2]-r12*q1[2]};
    float n2 = rsqrtf(u2[0]*u2[0] + u2[1]*u2[1] + u2[2]*u2[2] + 1e-12f);
    Q[0]=q0[0]; Q[1]=q1[0]; Q[2]=u2[0]*n2;
    Q[3]=q0[1]; Q[4]=q1[1]; Q[5]=u2[1]*n2;
    Q[6]=q0[2]; Q[7]=q1[2]; Q[8]=u2[2]*n2;
}

// gamma1 = QR(f^T x) per batch.  f3 ld 16.  grid=16, block=32
__global__ void qr1_kernel(const float* __restrict__ f3, const float* __restrict__ xc,
                           float* __restrict__ gamma1)
{
    int b = blockIdx.x, tid = threadIdx.x;
    __shared__ float part[32][9];
    float acc[9];
#pragma unroll
    for (int q = 0; q < 9; ++q) acc[q] = 0.f;
    for (int n = tid; n < NND; n += 32) {
        size_t row = (size_t)b * NND + n;
#pragma unroll
        for (int d = 0; d < 3; ++d)
#pragma unroll
            for (int e = 0; e < 3; ++e)
                acc[d * 3 + e] += f3[row * 16 + d] * xc[row * 3 + e];
    }
#pragma unroll
    for (int q = 0; q < 9; ++q) part[tid][q] = acc[q];
    __syncthreads();
    if (tid == 0) {
        float M[9];
#pragma unroll
        for (int q = 0; q < 9; ++q) {
            float s = 0.f;
            for (int t = 0; t < 32; ++t) s += part[t][q];
            M[q] = s;
        }
        qr3(M, gamma1 + b * 9);
    }
}

// g0 = QR(dec9) ; gamma = g0 @ gamma1^T.  dec9 ld 16.  grid=1, block=16
__global__ void qr2_kernel(const float* __restrict__ dec9, const float* __restrict__ gamma1,
                           float* __restrict__ gamma)
{
    int b = threadIdx.x;
    if (b >= BSZ) return;
    float g0[9];
    qr3(dec9 + b * 16, g0);
    const float* g1 = gamma1 + b * 9;
#pragma unroll
    for (int d = 0; d < 3; ++d)
#pragma unroll
        for (int f = 0; f < 3; ++f) {
            float s = 0.f;
#pragma unroll
            for (int e = 0; e < 3; ++e) s += g0[d * 3 + e] * g1[f * 3 + e];
            gamma[b * 9 + d * 3 + f] = s;
        }
}

// ---------------------------------------------------------------------------
// layernorm + adaLN modulate: out = ln(x)*(1+sc)+sh
// ---------------------------------------------------------------------------
__global__ void lnmod_kernel(const float* __restrict__ x, const float* __restrict__ mods,
                             int mld, int sh_off, int sc_off, float* __restrict__ out)
{
    int row = blockIdx.x, c = threadIdx.x, b = row >> 7;
    __shared__ float red[256];
    float v = x[(size_t)row * 256 + c];
    red[c] = v; __syncthreads();
    for (int s = 128; s > 0; s >>= 1) {
        if (c < s) red[c] += red[c + s];
        __syncthreads();
    }
    float mean = red[0] * (1.0f / 256.0f);
    __syncthreads();
    float dv = v - mean;
    red[c] = dv * dv; __syncthreads();
    for (int s = 128; s > 0; s >>= 1) {
        if (c < s) red[c] += red[c + s];
        __syncthreads();
    }
    float var = red[0] * (1.0f / 256.0f);
    float xn = dv * rsqrtf(var + 1e-6f);
    float sc = mods[(size_t)b * mld + sc_off + c];
    float sh = mods[(size_t)b * mld + sh_off + c];
    out[(size_t)row * 256 + c] = xn * (1.0f + sc) + sh;
}

// ---------------------------------------------------------------------------
// final: center xo, rotate by gamma^T, assemble output.  Y ld 16.
// ---------------------------------------------------------------------------
__global__ void final_kernel(const float* __restrict__ Y, const float* __restrict__ rowmask,
                             const float* __restrict__ Nn, const float* __restrict__ gamma,
                             float* __restrict__ out)
{
    int b = blockIdx.x, n = threadIdx.x, row = b * NND + n;
    __shared__ float red[NND];
    __shared__ float sums[3];
    float xo[3];
#pragma unroll
    for (int d = 0; d < 3; ++d) xo[d] = Y[(size_t)row * 16 + d];
    for (int d = 0; d < 3; ++d) {
        red[n] = xo[d]; __syncthreads();
        for (int s = 64; s > 0; s >>= 1) {
            if (n < s) red[n] += red[n + s];
            __syncthreads();
        }
        if (n == 0) sums[d] = red[0];
        __syncthreads();
    }
    float m = rowmask[row], inv = 1.0f / Nn[b];
    float xcn[3];
#pragma unroll
    for (int d = 0; d < 3; ++d) xcn[d] = xo[d] - sums[d] * inv * m;
    const float* g = gamma + b * 9;
#pragma unroll
    for (int e = 0; e < 3; ++e)
        out[(size_t)row * 9 + e] = xcn[0] * g[e * 3 + 0] + xcn[1] * g[e * 3 + 1] + xcn[2] * g[e * 3 + 2];
#pragma unroll
    for (int c = 3; c < 9; ++c)
        out[(size_t)row * 9 + c] = Y[(size_t)row * 16 + c];
}

// ---------------------------------------------------------------------------
// Host side
// ---------------------------------------------------------------------------
static inline void launch_gemm(hipStream_t s,
    const float* A, int lda, const float* B, int ldb,
    const float* bias, float bmult, float* C, int ldc,
    int M, int N, int K, int act,
    const float* rowscale, const float* rowmask,
    const float* gate, int gld, const float* residual, int ldr)
{
    dim3 grid((N + 63) / 64, M / 16);
    wmma_gemm_kernel<<<grid, 128, 0, s>>>(A, lda, B, ldb, bias, bmult, C, ldc,
                                          M, N, K, act, rowscale, rowmask, gate, gld,
                                          residual, ldr);
}

// Flattened input order (jax pytree: dict keys sorted alphabetically)
enum {
    IN_EDGE = 0, IN_MASK, IN_NOISE0, IN_NOISE1,
    P_ADA_B, P_ADA_W, P_MLP1_B, P_MLP1_W, P_MLP2_B, P_MLP2_W,
    P_PROJ_B, P_PROJ_W, P_QKV_B, P_QKV_W,
    P_DT1_B, P_DT1_W, P_DT2_B, P_DT2_W,
    P_FADA_B, P_FADA_W, P_FLIN_B, P_FLIN_W,
    P_G0D1_B, P_G0D1_W, P_G0D2_B, P_G0D2_W,
    P_G0E1_B, P_G0E1_W, P_G0E2_B, P_G0E2_W,
    P_G1MEANS, P_G1M1_B, P_G1M1_W, P_G1M2_B, P_G1M2_W, P_G1STDS,
    P_GKMEANS, P_GKSTDS,
    P_PG1_B, P_PG1_W, P_PK_B, P_PK_W,
    P_TM1_B, P_TM1_W, P_TM2_B, P_TM2_W,
    P_XHE_B, P_XHE_W,
    IN_T, IN_XH
};

extern "C" void kernel_launch(void* const* d_in, const int* in_sizes, int n_in,
                              void* d_out, int out_size, void* d_ws, size_t ws_size,
                              hipStream_t stream)
{
    (void)in_sizes; (void)n_in; (void)out_size; (void)ws_size;
    auto F = [&](int i) { return (const float*)d_in[i]; };

    const float* nmask  = F(IN_MASK);
    const float* noise0 = F(IN_NOISE0);
    const float* noise1 = F(IN_NOISE1);
    const float* tin    = F(IN_T);
    const float* xh     = F(IN_XH);
    float* dout = (float*)d_out;

    float* ws = (float*)d_ws;
    size_t off = 0;
    auto alloc = [&](size_t nfl) {
        nfl = (nfl + 15) & ~(size_t)15;      // 64B-align every buffer
        float* p = ws + off; off += nfl; return p;
    };

    float* xc      = alloc(ROWS * 3);
    float* rowmask = alloc(ROWS);
    float* rowinv  = alloc(ROWS);
    float* NnBuf   = alloc(BSZ);
    float* eemb    = alloc(BSZ * 256);
    float* th      = alloc(BSZ * 256);
    float* tvec    = alloc(BSZ * 256);
    float* ch      = alloc(BSZ * 256);
    float* cvec    = alloc(BSZ * 256);
    float* scact   = alloc(BSZ * 256);
    float* S1      = alloc((size_t)ROWS * 128);
    float* Sk      = alloc((size_t)ROWS * 128);
    float* f1      = alloc((size_t)ROWS * 64);
    float* fcat    = alloc((size_t)ROWS * 352);
    float* g1h     = alloc((size_t)ROWS * 256);
    float* f3      = alloc((size_t)ROWS * 16);
    float* gamma1  = alloc(BSZ * 9);
    float* gxcat   = alloc((size_t)ROWS * 288);
    float* ench    = alloc((size_t)ROWS * 256);
    float* enc     = alloc((size_t)ROWS * 256);
    float* gxmean  = alloc(BSZ * 256);
    float* dech    = alloc(BSZ * 256);
    float* dec9    = alloc(BSZ * 16);
    float* gammaB  = alloc(BSZ * 9);
    float* cxh     = alloc((size_t)ROWS * 32);
    float* xh2     = alloc((size_t)ROWS * 256);
    float* mods    = alloc(BSZ * 1536);
    float* hbuf    = alloc((size_t)ROWS * 256);
    float* qkvb    = alloc((size_t)ROWS * 768);
    float* attno   = alloc((size_t)ROWS * 256);
    float* mlpt    = alloc((size_t)ROWS * 1024);
    float* fm      = alloc(BSZ * 512);
    float* Yp      = alloc((size_t)ROWS * 16);
    // padded weight copies
    float* g1m1p   = alloc((size_t)352 * 256);
    float* g0e1p   = alloc((size_t)288 * 256);
    float* xhep    = alloc((size_t)32 * 192);
    float* g1m2p   = alloc((size_t)256 * 16);
    float* g0d2p   = alloc((size_t)256 * 16);
    float* flinp   = alloc((size_t)256 * 16);
    float* g1m2bp  = alloc(16);
    float* g0d2bp  = alloc(16);
    float* flinbp  = alloc(16);

    auto pad = [&](const float* src, int sr, int sc, float* dst, int dr, int dc) {
        int tot = dr * dc;
        pad_kernel<<<(tot + 255) / 256, 256, 0, stream>>>(src, sr, sc, dst, dr, dc);
    };
    pad(F(P_G1M1_W), 336, 256, g1m1p, 352, 256);
    pad(F(P_G0E1_W), 275, 256, g0e1p, 288, 256);
    pad(F(P_XHE_W),    9, 192, xhep,   32, 192);
    pad(F(P_G1M2_W), 256,   3, g1m2p, 256,  16);
    pad(F(P_G0D2_W), 256,   9, g0d2p, 256,  16);
    pad(F(P_FLIN_W), 256,   9, flinp, 256,  16);
    pad(F(P_G1M2_B),   1,   3, g1m2bp,  1,  16);
    pad(F(P_G0D2_B),   1,   9, g0d2bp,  1,  16);
    pad(F(P_FLIN_B),   1,   9, flinbp,  1,  16);

    // ---- preprocessing ----
    prep_kernel<<<BSZ, NND, 0, stream>>>(xh, nmask, xc, rowmask, rowinv, NnBuf);
    temb_kernel<<<BSZ, 256, 0, stream>>>(tin, eemb);

    // t_embedder (t_mlp) and DiT conditioner
    launch_gemm(stream, eemb, 256, F(P_TM1_W), 256, F(P_TM1_B), 1.f, th, 256,
                BSZ, 256, 256, 2, 0, 0, 0, 0, 0, 0);
    launch_gemm(stream, th, 256, F(P_TM2_W), 256, F(P_TM2_B), 1.f, tvec, 256,
                BSZ, 256, 256, 0, 0, 0, 0, 0, 0, 0);
    launch_gemm(stream, eemb, 256, F(P_DT1_W), 256, F(P_DT1_B), 1.f, ch, 256,
                BSZ, 256, 256, 2, 0, 0, 0, 0, 0, 0);
    launch_gemm(stream, ch, 256, F(P_DT2_W), 256, F(P_DT2_B), 1.f, cvec, 256,
                BSZ, 256, 256, 0, 0, 0, 0, 0, 0, 0);
    elem_silu_kernel<<<(BSZ * 256 + 255) / 256, 256, 0, stream>>>(cvec, scact, BSZ * 256);

    // gaussian layers (fused reductions) + projections
    gauss_kernel<<<ROWS, NND, 0, stream>>>(xc, rowmask, F(P_G1MEANS), F(P_G1STDS),
                                           F(P_GKMEANS), F(P_GKSTDS), S1, Sk);
    launch_gemm(stream, S1, 128, F(P_PG1_W), 64, F(P_PG1_B), 128.f, f1, 64,
                ROWS, 64, 128, 0, rowinv, 0, 0, 0, 0, 0);

    // g1 branch -> gamma_1
    fcat_kernel<<<ROWS, 128, 0, stream>>>(f1, noise1, tvec, rowmask, fcat);
    launch_gemm(stream, fcat, 352, g1m1p, 256, F(P_G1M1_B), 1.f, g1h, 256,
                ROWS, 256, 352, 1, 0, 0, 0, 0, 0, 0);
    launch_gemm(stream, g1h, 256, g1m2p, 16, g1m2bp, 1.f, f3, 16,
                ROWS, 16, 256, 0, 0, rowmask, 0, 0, 0, 0);
    qr1_kernel<<<BSZ, 32, 0, stream>>>(f3, xc, gamma1);

    // g0 branch -> gamma
    gxcat_kernel<<<ROWS, 128, 0, stream>>>(xc, gamma1, noise0, tvec, rowmask, gxcat);
    launch_gemm(stream, gxcat, 288, g0e1p, 256, F(P_G0E1_B), 1.f, ench, 256,
                ROWS, 256, 288, 1, 0, 0, 0, 0, 0, 0);
    launch_gemm(stream, ench, 256, F(P_G0E2_W), 256, F(P_G0E2_B), 1.f, enc, 256,
                ROWS, 256, 256, 0, 0, rowmask, 0, 0, 0, 0);
    colmean_kernel<<<BSZ, 256, 0, stream>>>(enc, NnBuf, gxmean);
    launch_gemm(stream, gxmean, 256, F(P_G0D1_W), 256, F(P_G0D1_B), 1.f, dech, 256,
                BSZ, 256, 256, 1, 0, 0, 0, 0, 0, 0);
    launch_gemm(stream, dech, 256, g0d2p, 16, g0d2bp, 1.f, dec9, 16,
                BSZ, 16, 256, 0, 0, 0, 0, 0, 0, 0);
    qr2_kernel<<<1, 16, 0, stream>>>(dec9, gamma1, gammaB);

    // DiT input: xh2 = mask * concat( [x@gamma, h]@xh_emb , pos_emb )
    cxh_kernel<<<ROWS, 32, 0, stream>>>(xc, gammaB, xh, cxh);
    launch_gemm(stream, cxh, 32, xhep, 192, F(P_XHE_B), 1.f, xh2, 256,
                ROWS, 192, 32, 0, 0, rowmask, 0, 0, 0, 0);
    launch_gemm(stream, Sk, 128, F(P_PK_W), 64, F(P_PK_B), 128.f, xh2 + 192, 256,
                ROWS, 64, 128, 0, rowinv, rowmask, 0, 0, 0, 0);

    // ---- DiT blocks ----
    for (int d = 0; d < 8; ++d) {
        const float* aw  = F(P_ADA_W)  + (size_t)d * 256 * 1536;
        const float* ab  = F(P_ADA_B)  + (size_t)d * 1536;
        const float* qw  = F(P_QKV_W)  + (size_t)d * 256 * 768;
        const float* qb  = F(P_QKV_B)  + (size_t)d * 768;
        const float* pw  = F(P_PROJ_W) + (size_t)d * 256 * 256;
        const float* pb  = F(P_PROJ_B) + (size_t)d * 256;
        const float* m1w = F(P_MLP1_W) + (size_t)d * 256 * 1024;
        const float* m1b = F(P_MLP1_B) + (size_t)d * 1024;
        const float* m2w = F(P_MLP2_W) + (size_t)d * 1024 * 256;
        const float* m2b = F(P_MLP2_B) + (size_t)d * 256;

        launch_gemm(stream, scact, 256, aw, 1536, ab, 1.f, mods, 1536,
                    BSZ, 1536, 256, 0, 0, 0, 0, 0, 0, 0);
        lnmod_kernel<<<ROWS, 256, 0, stream>>>(xh2, mods, 1536, 0, 256, hbuf);
        launch_gemm(stream, hbuf, 256, qw, 768, qb, 1.f, qkvb, 768,
                    ROWS, 768, 256, 0, 0, 0, 0, 0, 0, 0);
        attn_kernel<<<BSZ * 8, 128, 128 * 128 * 4 + 32 * 128 * 2, stream>>>(qkvb, nmask, attno);
        launch_gemm(stream, attno, 256, pw, 256, pb, 1.f, xh2, 256,
                    ROWS, 256, 256, 0, 0, 0, mods + 512, 1536, xh2, 256);
        lnmod_kernel<<<ROWS, 256, 0, stream>>>(xh2, mods, 1536, 768, 1024, hbuf);
        launch_gemm(stream, hbuf, 256, m1w, 1024, m1b, 1.f, mlpt, 1024,
                    ROWS, 1024, 256, 1, 0, 0, 0, 0, 0, 0);
        launch_gemm(stream, mlpt, 1024, m2w, 256, m2b, 1.f, xh2, 256,
                    ROWS, 256, 1024, 0, 0, 0, mods + 1280, 1536, xh2, 256);
    }

    // ---- DiT final layer ----
    launch_gemm(stream, scact, 256, F(P_FADA_W), 512, F(P_FADA_B), 1.f, fm, 512,
                BSZ, 512, 256, 0, 0, 0, 0, 0, 0, 0);
    lnmod_kernel<<<ROWS, 256, 0, stream>>>(xh2, fm, 512, 0, 256, hbuf);
    launch_gemm(stream, hbuf, 256, flinp, 16, flinbp, 1.f, Yp, 16,
                ROWS, 16, 256, 0, 0, rowmask, 0, 0, 0, 0);

    // ---- output assembly ----
    final_kernel<<<BSZ, NND, 0, stream>>>(Yp, rowmask, NnBuf, gammaB, dout);
}
